// MultiViewMultiPoseMHR_84937273246147
// MI455X (gfx1250) — matrix operand order repaired and checked by reading, not compiled, not run on hardware
//
#include <hip/hip_runtime.h>
#include <hip/hip_bf16.h>
#include <math.h>

// ---------------- sizes ----------------
#define VIEWS 2
#define NV    4000
#define NF    8000
#define NP    2500
#define HH    128
#define WW    128
#define NCOL  (NV*3)            // 12000
#define KTOT  340               // 64 + 72 + 204
#define KPAD  352               // 11 * 32
#define SIGMA_INV 1.0e5f
#define BLURF 9.210240372e-5f   // ln(9999) * 1e-5
#define ZNEARF 0.05f

typedef _Float16 v16h __attribute__((ext_vector_type(16)));
typedef float    v8f  __attribute__((ext_vector_type(8)));

// ---------------- small helpers ----------------
struct F3 { float x, y, z; };
__device__ __forceinline__ F3 f3sub(F3 a, F3 b) { return {a.x-b.x, a.y-b.y, a.z-b.z}; }
__device__ __forceinline__ float f3dot(F3 a, F3 b) { return a.x*b.x + a.y*b.y + a.z*b.z; }
__device__ __forceinline__ F3 f3mad(F3 a, F3 d, float t) { return {a.x+d.x*t, a.y+d.y*t, a.z+d.z*t}; }
__device__ __forceinline__ float clamp01(float t) { return fminf(fmaxf(t, 0.f), 1.f); }

// Exact translation of reference point_tri_sqdist3d (sequential where's)
__device__ float pt_tri_sqdist3d(F3 p, F3 a, F3 b, F3 c) {
  const float eps = 1e-12f;
  F3 ab = f3sub(b, a), ac = f3sub(c, a), ap = f3sub(p, a);
  float d1 = f3dot(ab, ap), d2 = f3dot(ac, ap);
  F3 bp = f3sub(p, b);
  float d3 = f3dot(ab, bp), d4 = f3dot(ac, bp);
  F3 cp = f3sub(p, c);
  float d5 = f3dot(ab, cp), d6 = f3dot(ac, cp);
  float va = d3*d6 - d5*d4;
  float vb = d5*d2 - d1*d6;
  float vc = d1*d4 - d3*d2;
  float den = va + vb + vc;
  den = (fabsf(den) > eps) ? den : 1.f;
  F3 cl = { a.x + ab.x*(vb/den) + ac.x*(vc/den),
            a.y + ab.y*(vb/den) + ac.y*(vc/den),
            a.z + ab.z*(vb/den) + ac.z*(vc/den) };
  float dab = (fabsf(d1-d3) > eps) ? (d1-d3) : 1.f;
  F3 cp_ab = f3mad(a, ab, clamp01(d1/dab));
  float dac = (fabsf(d2-d6) > eps) ? (d2-d6) : 1.f;
  F3 cp_ac = f3mad(a, ac, clamp01(d2/dac));
  float num_bc = d4 - d3;
  float den_bc = num_bc + (d5 - d6);
  float t_bc = clamp01(num_bc / ((fabsf(den_bc) > eps) ? den_bc : 1.f));
  F3 cb = f3sub(c, b);
  F3 cp_bc = f3mad(b, cb, t_bc);
  if (va <= 0.f && num_bc >= 0.f && (d5-d6) >= 0.f) cl = cp_bc;
  if (vb <= 0.f && d2 >= 0.f && d6 <= 0.f)          cl = cp_ac;
  if (vc <= 0.f && d1 >= 0.f && d3 <= 0.f)          cl = cp_ab;
  if (d6 >= 0.f && d5 <= d6)                        cl = c;
  if (d3 >= 0.f && d4 <= d3)                        cl = b;
  if (d1 <= 0.f && d2 <= 0.f)                       cl = a;
  F3 df = f3sub(p, cl);
  return f3dot(df, df);
}

__device__ __forceinline__ float seg_sq2(float qx, float qy, float ax, float ay,
                                         float bx, float by) {
  float abx = bx-ax, aby = by-ay;
  float t = clamp01(((qx-ax)*abx + (qy-ay)*aby) / fmaxf(abx*abx + aby*aby, 1e-12f));
  float dx = qx - (ax + t*abx), dy = qy - (ay + t*aby);
  return dx*dx + dy*dy;
}
__device__ __forceinline__ float signed_tri2(float qx, float qy, float ax, float ay,
                                             float bx, float by, float cx, float cy) {
  float d = fminf(fminf(seg_sq2(qx,qy,ax,ay,bx,by), seg_sq2(qx,qy,bx,by,cx,cy)),
                  seg_sq2(qx,qy,cx,cy,ax,ay));
  float s1 = (bx-ax)*(qy-ay) - (by-ay)*(qx-ax);
  float s2 = (cx-bx)*(qy-by) - (cy-by)*(qx-bx);
  float s3 = (ax-cx)*(qy-cy) - (ay-cy)*(qx-cx);
  bool inside = ((s1>=0.f)&&(s2>=0.f)&&(s3>=0.f)) || ((s1<=0.f)&&(s2<=0.f)&&(s3<=0.f));
  return inside ? -d : d;
}

// ---- branchless gathers for the WMMA fragments (no EXEC divergence) ----
// coefficient matrix C (16x352): rows 0..1 valid (views), rest zero
__device__ __forceinline__ float coef_val_bl(int m, int k, const float* poses,
                                             const float* idv, const float* exv) {
  int kc = (k < KTOT) ? k : (KTOT - 1);      // keep every candidate address in-bounds
  const float* p = idv + kc;                 // k in [0,64)
  p = (kc >= 64)  ? (exv + (kc - 64))                     : p;   // [64,136)
  p = (kc >= 136) ? (poses + (m & 1) * 204 + (kc - 136))  : p;   // [136,340)
  float v = *p;                              // single unconditional load
  return ((m < VIEWS) & (k < KTOT)) ? v : 0.f;
}
// stacked basis G (352x12000), rows beyond 340 zero
__device__ __forceinline__ float basis_val_bl(int k, int col, const float* sb,
                                              const float* eb, const float* pb) {
  int kc = (k < KTOT) ? k : (KTOT - 1);
  const float* p = sb + kc * NCOL;
  p = (kc >= 64)  ? (eb + (kc - 64)  * NCOL) : p;
  p = (kc >= 136) ? (pb + (kc - 136) * NCOL) : p;
  float v = p[col];
  return (k < KTOT) ? v : 0.f;
}

// ---------------- Kernel 1: blendshape GEMM via v_wmma_f32_16x16x32_f16 ----------------
// One wave per 16-column tile of the (2 x 12000) output; K = 352 in 11 unrolled
// steps of v_wmma_f32_16x16x32_f16. All fragment gathers are branchless.
__global__ void blend_wmma_k(const float* __restrict__ poses,
                             const float* __restrict__ idv,
                             const float* __restrict__ exv,
                             const float* __restrict__ ts,
                             const float* __restrict__ tmpl,
                             const float* __restrict__ sb,
                             const float* __restrict__ eb,
                             const float* __restrict__ pb,
                             float* __restrict__ verts) {
  int lane = threadIdx.x & 31;
  int tile = (blockIdx.x * blockDim.x + threadIdx.x) >> 5;   // 0..749
  int h  = lane >> 4;        // lane half
  int mn = lane & 15;        // A: M row / B: N column
  int col = tile * 16 + mn;

  v8f acc = {};
#pragma unroll
  for (int step = 0; step < KPAD/32; ++step) {
    int kk = step * 32;
    v16h af, bf;
    // A fragment: 16x32 f16, lane = M, lane-half selects K 0-15 / 16-31
#pragma unroll
    for (int j = 0; j < 8; ++j) {
      int kb = h*16 + 2*j;
      af[2*j]   = (_Float16)coef_val_bl(mn, kk + kb,     poses, idv, exv);
      af[2*j+1] = (_Float16)coef_val_bl(mn, kk + kb + 1, poses, idv, exv);
    }
    // B fragment: 32x16 f16, lane = N, lane-half selects K 0-15 / 16-31
#pragma unroll
    for (int j = 0; j < 8; ++j) {
      int k0 = kk + h*16 + 2*j;
      bf[2*j]   = (_Float16)basis_val_bl(k0,     col, sb, eb, pb);
      bf[2*j+1] = (_Float16)basis_val_bl(k0 + 1, col, sb, eb, pb);
    }
    acc = __builtin_amdgcn_wmma_f32_16x16x32_f16(false, af, false, bf,
                                                 (short)0, acc, false, false);
  }

  // D layout: lanes 0-15 hold rows M=0..7 in acc[0..7]; we need M=0,1 (the 2 views)
  if (lane < 16) {
    int d = col % 3;
    float sgn = (d == 0) ? 1.f : -1.f;       // * [1,-1,-1]
    float t0 = tmpl[col];
#pragma unroll
    for (int m = 0; m < VIEWS; ++m) {
      float raw = t0 + acc[m];
      verts[m*NCOL + col] = raw * 0.01f * sgn + ts[m*3 + d];
    }
  }
}

// ---------------- Kernel 2: projection ----------------
__global__ void project_k(const float* __restrict__ verts,
                          const float* __restrict__ fov,
                          float* __restrict__ proj, float* __restrict__ zb) {
  int t = blockIdx.x * blockDim.x + threadIdx.x;
  if (t >= VIEWS*NV) return;
  int view = t / NV, v = t % NV;
  const float* p = verts + view*NCOL + v*3;
  float cx = -p[0], cy = -p[1], cz = p[2];           // v_cam = verts * (-1,-1,1)
  float s = 1.f / tanf(fov[view] * 3.14159265358979f / 360.f);
  float iz = fmaxf(cz, 1e-6f);
  proj[(view*NV + v)*2 + 0] = s * cx / iz;
  proj[(view*NV + v)*2 + 1] = s * cy / iz;
  zb[view*NV + v] = cz;
}

// ---------------- Kernel 3: facemin (min over points per face) ----------------
__global__ void facemin_k(const float* __restrict__ verts,
                          const float* __restrict__ points,
                          const int* __restrict__ faces,
                          float* __restrict__ partials) {
  __shared__ float sp[NP*3];      // 30 KB of the 320 KB WGP LDS pool
  __shared__ float red[256];
  int view = blockIdx.y;
  for (int i = threadIdx.x; i < NP*3; i += blockDim.x)
    sp[i] = points[view*NP*3 + i];
  __syncthreads();

  int f = blockIdx.x * blockDim.x + threadIdx.x;
  float fm = 0.f;
  if (f < NF) {
    int i0 = faces[f*3+0], i1 = faces[f*3+1], i2 = faces[f*3+2];
    const float* vb = verts + view*NCOL;
    F3 a = { vb[i0*3], vb[i0*3+1], vb[i0*3+2] };
    F3 b = { vb[i1*3], vb[i1*3+1], vb[i1*3+2] };
    F3 c = { vb[i2*3], vb[i2*3+1], vb[i2*3+2] };
    float mn = 1e30f;
    for (int p = 0; p < NP; ++p) {
      F3 P = { sp[p*3], sp[p*3+1], sp[p*3+2] };
      mn = fminf(mn, pt_tri_sqdist3d(P, a, b, c));
    }
    fm = mn;
  }
  red[threadIdx.x] = fm;
  __syncthreads();
  for (int s = 128; s > 0; s >>= 1) {
    if (threadIdx.x < (unsigned)s) red[threadIdx.x] += red[threadIdx.x + s];
    __syncthreads();
  }
  if (threadIdx.x == 0) partials[view*32 + blockIdx.x] = red[0];
}

// ---------------- Kernel 4: minpt (min over faces per point) ----------------
__global__ void minpt_k(const float* __restrict__ verts,
                        const float* __restrict__ points,
                        const int* __restrict__ faces,
                        float* __restrict__ partials) {
  __shared__ float st[256*9];     // 9 KB triangle chunk
  __shared__ float red[256];
  int view = blockIdx.y;
  int p = blockIdx.x * blockDim.x + threadIdx.x;
  bool act = (p < NP);
  F3 P = {0.f, 0.f, 0.f};
  if (act) { P.x = points[view*NP*3 + p*3];
             P.y = points[view*NP*3 + p*3+1];
             P.z = points[view*NP*3 + p*3+2]; }
  float mn = 1e30f;
  const float* vb = verts + view*NCOL;
  for (int base = 0; base < NF; base += 256) {
    int f = base + threadIdx.x;
    if (f < NF) {
      int i0 = faces[f*3+0], i1 = faces[f*3+1], i2 = faces[f*3+2];
      float* s = &st[threadIdx.x*9];
      s[0]=vb[i0*3]; s[1]=vb[i0*3+1]; s[2]=vb[i0*3+2];
      s[3]=vb[i1*3]; s[4]=vb[i1*3+1]; s[5]=vb[i1*3+2];
      s[6]=vb[i2*3]; s[7]=vb[i2*3+1]; s[8]=vb[i2*3+2];
    }
    __syncthreads();
    int cnt = min(256, NF - base);
    for (int t = 0; t < cnt; ++t) {
      const float* s = &st[t*9];
      F3 a = {s[0],s[1],s[2]}, b = {s[3],s[4],s[5]}, c = {s[6],s[7],s[8]};
      mn = fminf(mn, pt_tri_sqdist3d(P, a, b, c));
    }
    __syncthreads();
  }
  red[threadIdx.x] = act ? mn : 0.f;
  __syncthreads();
  for (int s = 128; s > 0; s >>= 1) {
    if (threadIdx.x < (unsigned)s) red[threadIdx.x] += red[threadIdx.x + s];
    __syncthreads();
  }
  if (threadIdx.x == 0) partials[view*10 + blockIdx.x] = red[0];
}

// ---------------- Kernel 5: soft raster scan ----------------
__global__ void raster_k(const float* __restrict__ proj,
                         const float* __restrict__ zb,
                         const int* __restrict__ faces,
                         const float* __restrict__ masks,
                         float* __restrict__ alpha_out,
                         float* __restrict__ partials) {
  __shared__ float st[128*9];     // 4.6 KB: 128 tris x (3 verts 2D + 3 z)
  __shared__ float red[256];
  int view = blockIdx.y;
  int i = blockIdx.x * blockDim.x + threadIdx.x;   // 0..16383
  int py = i >> 7, px = i & 127;
  float gy = 1.f - (2.f*(float)py + 1.f) / (float)HH;
  float gx = 1.f - (2.f*(float)px + 1.f) / (float)WW;
  const float* pv = proj + view*NV*2;
  const float* zv = zb + view*NV;
  float acc = 0.f;

  for (int base = 0; base < NF; base += 128) {
    if (threadIdx.x < 128) {
      int f = base + threadIdx.x;
      if (f < NF) {
        int i0 = faces[f*3+0], i1 = faces[f*3+1], i2 = faces[f*3+2];
        float* s = &st[threadIdx.x*9];
        s[0]=pv[i0*2]; s[1]=pv[i0*2+1];
        s[2]=pv[i1*2]; s[3]=pv[i1*2+1];
        s[4]=pv[i2*2]; s[5]=pv[i2*2+1];
        s[6]=zv[i0];   s[7]=zv[i1];    s[8]=zv[i2];
      }
      // stream next chunk of indices through L2 (global_prefetch_b8)
      if (base + 128 < NF) __builtin_prefetch(&faces[(base+128)*3 + threadIdx.x*3], 0, 1);
    }
    __syncthreads();
    int cnt = min(128, NF - base);
    for (int t = 0; t < cnt; ++t) {
      const float* s = &st[t*9];
      float d = signed_tri2(gx, gy, s[0], s[1], s[2], s[3], s[4], s[5]);
      bool valid = (s[6] > ZNEARF) && (s[7] > ZNEARF) && (s[8] > ZNEARF) && (d < BLURF);
      if (valid) {
        float pr = 1.f / (1.f + __expf(d * SIGMA_INV));   // sigmoid(-d/sigma)
        pr = fminf(pr, 1.f - 1e-7f);
        acc += log1pf(-pr);
      }
    }
    __syncthreads();
  }

  float alpha = 1.f - __expf(acc);
  alpha_out[view*HH*WW + i] = alpha;
  float diff = alpha - masks[view*HH*WW + i];
  red[threadIdx.x] = diff * diff;
  __syncthreads();
  for (int s = 128; s > 0; s >>= 1) {
    if (threadIdx.x < (unsigned)s) red[threadIdx.x] += red[threadIdx.x + s];
    __syncthreads();
  }
  if (threadIdx.x == 0) partials[view*64 + blockIdx.x] = red[0];
}

// ---------------- Kernel 6: deterministic finalize ----------------
__global__ void finalize_k(const float* __restrict__ fm_part,
                           const float* __restrict__ mp_part,
                           const float* __restrict__ mk_part,
                           float* __restrict__ out2) {
  if (threadIdx.x == 0 && blockIdx.x == 0) {
    float fs = 0.f; for (int i = 0; i < 64;  ++i) fs += fm_part[i];
    float ms = 0.f; for (int i = 0; i < 20;  ++i) ms += mp_part[i];
    float ks = 0.f; for (int i = 0; i < 128; ++i) ks += mk_part[i];
    out2[0] = ms / (float)(VIEWS*NP) + fs / (float)(VIEWS*NF);   // pcl_loss
    out2[1] = ks / (float)(VIEWS*HH*WW);                         // mask_loss
  }
}

// ---------------- launch ----------------
extern "C" void kernel_launch(void* const* d_in, const int* in_sizes, int n_in,
                              void* d_out, int out_size, void* d_ws, size_t ws_size,
                              hipStream_t stream) {
  (void)in_sizes; (void)n_in; (void)out_size; (void)ws_size;
  const float* poses = (const float*)d_in[0];
  const float* idv   = (const float*)d_in[1];
  const float* exv   = (const float*)d_in[2];
  const float* ts    = (const float*)d_in[3];
  const float* pts   = (const float*)d_in[4];
  const float* masks = (const float*)d_in[5];
  const float* tmpl  = (const float*)d_in[6];
  const float* sb    = (const float*)d_in[7];
  const float* eb    = (const float*)d_in[8];
  const float* pb    = (const float*)d_in[9];
  const int*   faces = (const int*)d_in[10];
  const float* fov   = (const float*)d_in[11];

  float* out   = (float*)d_out;
  float* verts = out;                   // 2*4000*3 = 24000
  float* alpha = out + VIEWS*NCOL;      // 2*128*128 = 32768
  float* loss2 = out + VIEWS*NCOL + VIEWS*HH*WW;  // pcl_loss, mask_loss

  float* wsf     = (float*)d_ws;
  float* proj    = wsf;                 // 16000
  float* zbuf    = wsf + 16000;         // 8000
  float* fm_part = wsf + 24000;         // 64
  float* mp_part = wsf + 24064;         // 20
  float* mk_part = wsf + 24084;         // 128

  // 750 column tiles, 1 wave (32 lanes) each -> 375 blocks of 64 threads
  blend_wmma_k<<<dim3(375), dim3(64), 0, stream>>>(poses, idv, exv, ts, tmpl,
                                                   sb, eb, pb, verts);
  project_k<<<dim3((VIEWS*NV + 255)/256), dim3(256), 0, stream>>>(verts, fov, proj, zbuf);
  facemin_k<<<dim3(32, VIEWS), dim3(256), 0, stream>>>(verts, pts, faces, fm_part);
  minpt_k  <<<dim3(10, VIEWS), dim3(256), 0, stream>>>(verts, pts, faces, mp_part);
  raster_k <<<dim3(64, VIEWS), dim3(256), 0, stream>>>(proj, zbuf, faces, masks,
                                                       alpha, mk_part);
  finalize_k<<<dim3(1), dim3(32), 0, stream>>>(fm_part, mp_part, mk_part, loss2);
}
